// _RPN_distil_20873541059188
// MI455X (gfx1250) — matrix operand order repaired and checked by reading, not compiled, not run on hardware
//
#include <hip/hip_runtime.h>
#include <cstdint>
#include <cstddef>

// Flip to 0 if the gfx1250 assembler rejects the async mnemonics.
#define USE_ASYNC_LDS 1

// ---------- types ----------
typedef unsigned short u16t;
typedef __attribute__((ext_vector_type(16))) __bf16         v16bf;
typedef __attribute__((ext_vector_type(8)))  float          v8f;
typedef __attribute__((ext_vector_type(8)))  unsigned short us8;
typedef __attribute__((ext_vector_type(16))) unsigned short us16;
typedef __attribute__((ext_vector_type(4)))  unsigned int   ui4;

#define NEG_BIG (-1.0e10f)

// ---------- problem constants ----------
#define BATCH   4
#define CI      1024
#define CO      512
#define HDIM    64
#define WDIM    64
#define HW      (HDIM*WDIM)          // 4096
#define MTOT    (BATCH*HW)           // 16384
#define NANCH   9
#define NPROP   (HW*NANCH)           // 36864 per batch
#define PRE_N   6000
#define POST_N  300
#define HEADN   64                   // 18 cls + 36 bbox + 10 pad

// ---------- workspace layout (bytes) ----------
#define OFF_FEAT   ((size_t)0)                       // [B][H][W][CI] bf16  : 32 MB
#define OFF_WGT    ((size_t)33554432)                // [9][CO][CI]   bf16  : 9 MB
#define OFF_X      ((size_t)42991616)                // [B][H][W][CO] bf16  : 16 MB
#define OFF_HWGT   ((size_t)59768832)                // [64][512]     bf16  : 64 KB
#define OFF_HOUT   ((size_t)59834368)                // [M][64]       f32   : 4 MB
#define OFF_PROPS  ((size_t)64028672)                // [B][NPROP][4] f32
#define OFF_SCORES ((size_t)66387968)                // [B][NPROP]    f32
#define OFF_HIST   ((size_t)66977792)                // [B][2048]     i32
#define OFF_CNT    ((size_t)67010560)                // 16 i32: cntA[4] cntB[4] above[4] tbin[4]
#define OFF_SELS   ((size_t)67010816)                // [B][6000]     f32
#define OFF_SELB   ((size_t)67106816)                // [B][6000][4]  f32

// ---------- helpers ----------
static __device__ __forceinline__ u16t f2bf(float x) {
    unsigned u = __builtin_bit_cast(unsigned, x);
    unsigned r = (u + 0x7FFFu + ((u >> 16) & 1u)) >> 16;   // round-to-nearest-even
    return (u16t)r;
}
static __device__ __forceinline__ v8f zero8() {
    v8f z = {0.f,0.f,0.f,0.f,0.f,0.f,0.f,0.f};
    return z;
}
// build a 16xbf16 fragment from two 16B chunks
static __device__ __forceinline__ v16bf ld_frag(const u16t* p0, const u16t* p1) {
    us8 lo = *(const us8*)p0;
    us8 hi = *(const us8*)p1;
    us16 c = __builtin_shufflevector(lo, hi, 0,1,2,3,4,5,6,7,8,9,10,11,12,13,14,15);
    return __builtin_bit_cast(v16bf, c);
}

// ======================================================================
// 1) repack base_feat NCHW f32 -> NHWC bf16 (LDS tiled transpose)
// ======================================================================
__global__ __launch_bounds__(1024) void k_repack_feat(const float* __restrict__ in,
                                                      u16t* __restrict__ out) {
    __shared__ u16t tile[32][33];
    const int tx = threadIdx.x, ty = threadIdx.y;
    const int hw0 = blockIdx.x * 32, c0 = blockIdx.y * 32, b = blockIdx.z;
    tile[ty][tx] = f2bf(in[((size_t)(b * CI + c0 + ty)) * HW + hw0 + tx]);
    __syncthreads();
    out[((size_t)(b * HW + hw0 + ty)) * CI + c0 + tx] = tile[tx][ty];
}

// ======================================================================
// 2) repack W_conv OIHW f32 -> [tap][co][ci] bf16
// ======================================================================
__global__ __launch_bounds__(256) void k_repack_wgt(const float* __restrict__ in,
                                                    u16t* __restrict__ out) {
    const int i = blockIdx.x * 256 + threadIdx.x;             // 9*512*1024
    const int kpos = i / (CO * CI);
    const int rem  = i - kpos * (CO * CI);
    const int co = rem / CI, ci = rem - co * CI;
    out[i] = f2bf(in[((size_t)(co * CI + ci)) * 9 + kpos]);
}

// ======================================================================
// 3) repack head weights: rows 0..17 = W_cls, 18..53 = W_bbox, pad 0
// ======================================================================
__global__ __launch_bounds__(256) void k_repack_hwgt(const float* __restrict__ wc,
                                                     const float* __restrict__ wb,
                                                     u16t* __restrict__ out) {
    const int i = blockIdx.x * 256 + threadIdx.x;             // 64*512
    const int co = i >> 9, ci = i & 511;
    float v = 0.f;
    if (co < 18)      v = wc[co * 512 + ci];
    else if (co < 54) v = wb[(co - 18) * 512 + ci];
    out[i] = f2bf(v);
}

// ======================================================================
// 4) 3x3 conv as implicit GEMM on WMMA bf16.
//    block = 256 thr (8 wave32); tile 256M x 64N; each wave owns 32M x 64N
//    (2 A-frags x 4 B-frags = 8 WMMAs per 32-deep K-step).
//    Staging uses CDNA5 async global->LDS DMA (ASYNCcnt) when enabled.
//    grid (CO/64, MTOT/256)
// ======================================================================
__global__ __launch_bounds__(256) void k_conv3x3_wmma(const u16t* __restrict__ feat,
                                                      const u16t* __restrict__ wgt,
                                                      const float* __restrict__ bias,
                                                      u16t* __restrict__ xout) {
    __shared__ u16t Alds[256 * 32];   // [m][k]  16 KB
    __shared__ u16t Blds[64 * 32];    // [n][k]   4 KB
    const int tid  = threadIdx.x;
    const int lane = tid & 31, wave = tid >> 5;
    const int n0 = blockIdx.x * 64;
    const int m0 = blockIdx.y * 256;

    // staging roles: A = 1 row (64B) per thread; B = 16B per thread
    const int arow = tid;
    const int brow = tid >> 2, bchunk = tid & 3;

    const int am = m0 + arow;
    const int ab = am >> 12;
    const int ahw = am & 4095;
    const int ah = ahw >> 6, aw = ahw & 63;

    // fragment addressing (ISA 7.12.2)
    const int fr  = lane & 15;
    const int ak0 = (lane < 16) ? 0 : 16;
    const u16t* aPlo = &Alds[(wave * 32 + fr) * 32 + ak0];
    const u16t* aPhi = aPlo + 16 * 32;
    const int bk0 = (lane < 16) ? 0 : 16;
    const u16t* bP0 = &Blds[(0 * 16 + fr) * 32 + bk0];
    const u16t* bP1 = &Blds[(1 * 16 + fr) * 32 + bk0];
    const u16t* bP2 = &Blds[(2 * 16 + fr) * 32 + bk0];
    const u16t* bP3 = &Blds[(3 * 16 + fr) * 32 + bk0];

#if USE_ASYNC_LDS
    // low 32 bits of a generic pointer to LDS == LDS byte address (ISA 10.2)
    const unsigned laA = (unsigned)(uintptr_t)(&Alds[arow * 32]);
    const unsigned laB = (unsigned)(uintptr_t)(&Blds[brow * 32 + bchunk * 8]);
#endif

    v8f accL[4], accH[4];
#pragma unroll
    for (int g = 0; g < 4; ++g) { accL[g] = zero8(); accH[g] = zero8(); }

    for (int kpos = 0; kpos < 9; ++kpos) {
        const int dh = kpos / 3 - 1, dw = kpos % 3 - 1;
        const int ih = ah + dh, iw = aw + dw;
        const bool inb = ((unsigned)ih < (unsigned)HDIM) && ((unsigned)iw < (unsigned)WDIM);
        const int ihc = ih < 0 ? 0 : (ih > 63 ? 63 : ih);
        const int iwc = iw < 0 ? 0 : (iw > 63 ? 63 : iw);
        const u16t* asrc = feat + ((size_t)((ab * 64 + ihc) * 64 + iwc)) * CI;
        const u16t* bsrc = wgt + ((size_t)(kpos * CO + n0 + brow)) * CI + bchunk * 8;

        for (int kc = 0; kc < CI; kc += 32) {
#if USE_ASYNC_LDS
            // ---- stage tiles with async global->LDS DMA (offset applies to both sides) ----
            if (inb) {
                const u16t* ga = asrc + kc;
                asm volatile("global_load_async_to_lds_b128 %0, %1, off"
                             :: "v"(laA), "v"(ga) : "memory");
                asm volatile("global_load_async_to_lds_b128 %0, %1, off offset:16"
                             :: "v"(laA), "v"(ga) : "memory");
                asm volatile("global_load_async_to_lds_b128 %0, %1, off offset:32"
                             :: "v"(laA), "v"(ga) : "memory");
                asm volatile("global_load_async_to_lds_b128 %0, %1, off offset:48"
                             :: "v"(laA), "v"(ga) : "memory");
            } else {
                const ui4 z = {0u,0u,0u,0u};
                ui4* adst = (ui4*)&Alds[arow * 32];
                adst[0] = z; adst[1] = z; adst[2] = z; adst[3] = z;
            }
            {
                const u16t* gb = bsrc + kc;
                asm volatile("global_load_async_to_lds_b128 %0, %1, off"
                             :: "v"(laB), "v"(gb) : "memory");
            }
            asm volatile("s_wait_asynccnt 0x0" ::: "memory");
#else
            // ---- fallback: VGPR round-trip staging ----
            ui4 a0 = {0u,0u,0u,0u}, a1 = a0, a2 = a0, a3 = a0;
            if (inb) {
                const ui4* p = (const ui4*)(asrc + kc);
                a0 = p[0]; a1 = p[1]; a2 = p[2]; a3 = p[3];
            }
            ui4* adst = (ui4*)&Alds[arow * 32];
            adst[0] = a0; adst[1] = a1; adst[2] = a2; adst[3] = a3;
            *(ui4*)&Blds[brow * 32 + bchunk * 8] = *(const ui4*)(bsrc + kc);
#endif
            __syncthreads();

            const v16bf afL = ld_frag(aPlo, aPlo + 8);
            const v16bf afH = ld_frag(aPhi, aPhi + 8);
            const v16bf b0 = ld_frag(bP0, bP0 + 8);
            const v16bf b1 = ld_frag(bP1, bP1 + 8);
            const v16bf b2 = ld_frag(bP2, bP2 + 8);
            const v16bf b3 = ld_frag(bP3, bP3 + 8);
            accL[0] = __builtin_amdgcn_wmma_f32_16x16x32_bf16(false, afL, false, b0, (short)0, accL[0], false, false);
            accH[0] = __builtin_amdgcn_wmma_f32_16x16x32_bf16(false, afH, false, b0, (short)0, accH[0], false, false);
            accL[1] = __builtin_amdgcn_wmma_f32_16x16x32_bf16(false, afL, false, b1, (short)0, accL[1], false, false);
            accH[1] = __builtin_amdgcn_wmma_f32_16x16x32_bf16(false, afH, false, b1, (short)0, accH[1], false, false);
            accL[2] = __builtin_amdgcn_wmma_f32_16x16x32_bf16(false, afL, false, b2, (short)0, accL[2], false, false);
            accH[2] = __builtin_amdgcn_wmma_f32_16x16x32_bf16(false, afH, false, b2, (short)0, accH[2], false, false);
            accL[3] = __builtin_amdgcn_wmma_f32_16x16x32_bf16(false, afL, false, b3, (short)0, accL[3], false, false);
            accH[3] = __builtin_amdgcn_wmma_f32_16x16x32_bf16(false, afH, false, b3, (short)0, accH[3], false, false);
            __syncthreads();
        }
    }

    // epilogue: +bias, ReLU, store NHWC bf16
    const int mlo = m0 + wave * 32 + ((lane < 16) ? 0 : 8);
#pragma unroll
    for (int g = 0; g < 4; ++g) {
        const int n = n0 + g * 16 + fr;
        const float bv = bias[n];
#pragma unroll
        for (int r = 0; r < 8; ++r) {
            float v0 = accL[g][r] + bv;
            float v1 = accH[g][r] + bv;
            v0 = v0 > 0.f ? v0 : 0.f;
            v1 = v1 > 0.f ? v1 : 0.f;
            xout[((size_t)(mlo + r)) * CO + n]      = f2bf(v0);
            xout[((size_t)(mlo + 16 + r)) * CO + n] = f2bf(v1);
        }
    }
}

// ======================================================================
// 5) head GEMM: M=16384, K=512, N=64 (cls 18 | bbox 36 | pad), WMMA bf16
// ======================================================================
__global__ __launch_bounds__(256) void k_head_wmma(const u16t* __restrict__ x,
                                                   const u16t* __restrict__ hwgt,
                                                   const float* __restrict__ bcls,
                                                   const float* __restrict__ bbox_b,
                                                   float* __restrict__ hout) {
    __shared__ u16t Alds[128 * 32];
    __shared__ u16t Blds[64 * 32];
    const int tid  = threadIdx.x;
    const int lane = tid & 31, wave = tid >> 5;
    const int m0 = blockIdx.x * 128;

    const int arow = tid >> 1, apart = tid & 1;
    const int brow = tid >> 2, bchunk = tid & 3;

    const int fr  = lane & 15;
    const int ak0 = (lane < 16) ? 0 : 16;
    const u16t* aP = &Alds[(wave * 16 + fr) * 32 + ak0];
    const int bk0 = (lane < 16) ? 0 : 16;
    const u16t* bP0 = &Blds[(0 * 16 + fr) * 32 + bk0];
    const u16t* bP1 = &Blds[(1 * 16 + fr) * 32 + bk0];
    const u16t* bP2 = &Blds[(2 * 16 + fr) * 32 + bk0];
    const u16t* bP3 = &Blds[(3 * 16 + fr) * 32 + bk0];

    v8f acc[4];
#pragma unroll
    for (int g = 0; g < 4; ++g) acc[g] = zero8();

    const u16t* asrc = x + ((size_t)(m0 + arow)) * CO + apart * 16;
    const u16t* bsrc = hwgt + ((size_t)brow) * CO + bchunk * 8;

    for (int kc = 0; kc < CO; kc += 32) {
        ui4* adst = (ui4*)&Alds[arow * 32 + apart * 16];
        const ui4* ap = (const ui4*)(asrc + kc);
        adst[0] = ap[0]; adst[1] = ap[1];
        *(ui4*)&Blds[brow * 32 + bchunk * 8] = *(const ui4*)(bsrc + kc);
        __syncthreads();

        const v16bf af = ld_frag(aP, aP + 8);
        const v16bf b0 = ld_frag(bP0, bP0 + 8);
        const v16bf b1 = ld_frag(bP1, bP1 + 8);
        const v16bf b2 = ld_frag(bP2, bP2 + 8);
        const v16bf b3 = ld_frag(bP3, bP3 + 8);
        acc[0] = __builtin_amdgcn_wmma_f32_16x16x32_bf16(false, af, false, b0, (short)0, acc[0], false, false);
        acc[1] = __builtin_amdgcn_wmma_f32_16x16x32_bf16(false, af, false, b1, (short)0, acc[1], false, false);
        acc[2] = __builtin_amdgcn_wmma_f32_16x16x32_bf16(false, af, false, b2, (short)0, acc[2], false, false);
        acc[3] = __builtin_amdgcn_wmma_f32_16x16x32_bf16(false, af, false, b3, (short)0, acc[3], false, false);
        __syncthreads();
    }

    const int mrow = m0 + wave * 16 + ((lane < 16) ? 0 : 8);
#pragma unroll
    for (int g = 0; g < 4; ++g) {
        const int n = g * 16 + fr;
        float bv = 0.f;
        if (n < 18)      bv = bcls[n];
        else if (n < 54) bv = bbox_b[n - 18];
#pragma unroll
        for (int r = 0; r < 8; ++r) {
            hout[((size_t)(mrow + r)) * HEADN + n] = acc[g][r] + bv;
        }
    }
}

// ======================================================================
// 6) proposals: softmax-fg score, anchors, bbox_transform_inv, clip
// ======================================================================
__global__ __launch_bounds__(256) void k_prep(const float* __restrict__ hout,
                                              const float* __restrict__ im_info,
                                              float* __restrict__ props,
                                              float* __restrict__ scores) {
    const int i = blockIdx.x * 256 + threadIdx.x;             // BATCH*NPROP
    const int b  = i / NPROP;
    const int r  = i - b * NPROP;
    const int hw = r / NANCH;
    const int a  = r - hw * NANCH;
    const int h = hw >> 6, w = hw & 63;

    const float* row = hout + ((size_t)(b * HW + hw)) * HEADN;
    const float c0 = row[a], c1 = row[9 + a];
    const float score = 1.f / (1.f + __expf(c0 - c1));

    const float ratios[3] = {0.5f, 1.f, 2.f};
    const float scales[3] = {8.f, 16.f, 32.f};
    const float ratio = ratios[a / 3];
    const float scale = scales[a % 3];
    const float ws0 = rintf(sqrtf(256.f / ratio));
    const float hs0 = rintf(ws0 * ratio);
    const float wsa = ws0 * scale, hsa = hs0 * scale;
    const float cx = 7.5f + (float)w * 16.f;
    const float cy = 7.5f + (float)h * 16.f;
    const float ax1 = cx - 0.5f * (wsa - 1.f), ay1 = cy - 0.5f * (hsa - 1.f);
    const float ax2 = cx + 0.5f * (wsa - 1.f), ay2 = cy + 0.5f * (hsa - 1.f);

    const float aw_ = ax2 - ax1 + 1.f, ah_ = ay2 - ay1 + 1.f;
    const float acx = ax1 + 0.5f * aw_, acy = ay1 + 0.5f * ah_;
    const float d0 = row[18 + a * 4 + 0], d1 = row[18 + a * 4 + 1];
    const float d2 = row[18 + a * 4 + 2], d3 = row[18 + a * 4 + 3];
    const float pcx = d0 * aw_ + acx, pcy = d1 * ah_ + acy;
    const float pw = __expf(d2) * aw_, ph = __expf(d3) * ah_;

    const float xmax = im_info[b * 3 + 1] - 1.f;
    const float ymax = im_info[b * 3 + 0] - 1.f;
    float px1 = fminf(fmaxf(pcx - 0.5f * pw, 0.f), xmax);
    float py1 = fminf(fmaxf(pcy - 0.5f * ph, 0.f), ymax);
    float px2 = fminf(fmaxf(pcx + 0.5f * pw, 0.f), xmax);
    float py2 = fminf(fmaxf(pcy + 0.5f * ph, 0.f), ymax);

    float* pr = props + (size_t)i * 4;
    pr[0] = px1; pr[1] = py1; pr[2] = px2; pr[3] = py2;
    scores[i] = score;
}

// ======================================================================
// 7) init: pad selection buffers, zero hist + counters
// ======================================================================
__global__ __launch_bounds__(256) void k_init_select(float* __restrict__ sels,
                                                     float* __restrict__ selb,
                                                     int* __restrict__ hist,
                                                     int* __restrict__ cnt) {
    const int i = blockIdx.x * 256 + threadIdx.x;
    if (i < BATCH * PRE_N)      sels[i] = NEG_BIG;
    if (i < BATCH * PRE_N * 4)  selb[i] = 0.f;
    if (i < BATCH * 2048)       hist[i] = 0;
    if (i < 16)                 cnt[i]  = 0;
}

// 8) score histogram on top 11 float bits (scores are positive)
__global__ __launch_bounds__(256) void k_hist(const float* __restrict__ scores,
                                              int* __restrict__ hist) {
    const int i = blockIdx.x * 256 + threadIdx.x;
    const int b = i / NPROP;
    const unsigned key = __builtin_bit_cast(unsigned, scores[i]);
    atomicAdd(&hist[b * 2048 + (key >> 21)], 1);
}

// 9) find threshold bin (6000th largest) per batch
__global__ void k_thresh(const int* __restrict__ hist, int* __restrict__ cnt) {
    if (threadIdx.x != 0) return;
    const int b = blockIdx.x;
    int cum = 0, t = 0, above = 0;
    for (int bin = 2047; bin >= 0; --bin) {
        const int c = hist[b * 2048 + bin];
        if (cum + c >= PRE_N) { t = bin; above = cum; break; }
        cum += c;
    }
    cnt[8 + b]  = above;
    cnt[12 + b] = t;
}

// 10) compact top-6000 (all strictly-above bins, then boundary bin fills)
__global__ __launch_bounds__(256) void k_compact(const float* __restrict__ scores,
                                                 const float* __restrict__ props,
                                                 int* __restrict__ cnt,
                                                 float* __restrict__ sels,
                                                 float* __restrict__ selb) {
    const int i = blockIdx.x * 256 + threadIdx.x;
    const int b = i / NPROP;
    const float s = scores[i];
    const unsigned bin = __builtin_bit_cast(unsigned, s) >> 21;
    const int t = cnt[12 + b];
    int p = -1;
    if ((int)bin > t) {
        p = atomicAdd(&cnt[b], 1);
    } else if ((int)bin == t) {
        p = cnt[8 + b] + atomicAdd(&cnt[4 + b], 1);
        if (p >= PRE_N) p = -1;
    }
    if (p >= 0) {
        const size_t d = (size_t)b * PRE_N + p;
        sels[d] = s;
        const float* pr = props + (size_t)i * 4;
        float* o = selb + d * 4;
        o[0] = pr[0]; o[1] = pr[1]; o[2] = pr[2]; o[3] = pr[3];
    }
}

// ======================================================================
// 11) NMS: one workgroup per batch, 300 sequential argmax+suppress rounds
// ======================================================================
__global__ __launch_bounds__(1024) void k_nms(float* __restrict__ sels,
                                              const float* __restrict__ selb,
                                              float* __restrict__ out) {
    __shared__ float svals[1024];
    __shared__ int   sidx[1024];
    __shared__ float sbox[4];
    __shared__ int   sj, svalid;
    const int b = blockIdx.x, tid = threadIdx.x;
    float* sc = sels + (size_t)b * PRE_N;
    const float* bx = selb + (size_t)b * PRE_N * 4;
    float* rois = out + (size_t)b * POST_N * 5;

    if (b == 0 && tid == 0) {                 // tuple tail: two scalar zeros
        out[BATCH * POST_N * 5 + 0] = 0.f;
        out[BATCH * POST_N * 5 + 1] = 0.f;
    }

    for (int it = 0; it < POST_N; ++it) {
        float best = -3.0e38f; int bj = 0;
        for (int j = tid; j < PRE_N; j += 1024) {
            const float v = sc[j];
            if (v > best) { best = v; bj = j; }
        }
        svals[tid] = best; sidx[tid] = bj;
        __syncthreads();
        for (int s = 512; s > 0; s >>= 1) {
            if (tid < s && svals[tid + s] > svals[tid]) {
                svals[tid] = svals[tid + s];
                sidx[tid]  = sidx[tid + s];
            }
            __syncthreads();
        }
        if (tid == 0) {
            const int j = sidx[0];
            const int valid = (svals[0] > NEG_BIG * 0.5f);
            sj = j; svalid = valid;
            float b0 = 0.f, b1 = 0.f, b2 = 0.f, b3 = 0.f;
            if (valid) { b0 = bx[j*4]; b1 = bx[j*4+1]; b2 = bx[j*4+2]; b3 = bx[j*4+3]; }
            sbox[0] = b0; sbox[1] = b1; sbox[2] = b2; sbox[3] = b3;
            rois[it * 5 + 0] = (float)b;
            rois[it * 5 + 1] = b0; rois[it * 5 + 2] = b1;
            rois[it * 5 + 3] = b2; rois[it * 5 + 4] = b3;
        }
        __syncthreads();
        if (svalid) {
            const float bb0 = sbox[0], bb1 = sbox[1], bb2 = sbox[2], bb3 = sbox[3];
            const float barea = (bb2 - bb0 + 1.f) * (bb3 - bb1 + 1.f);
            const int jm = sj;
            for (int j = tid; j < PRE_N; j += 1024) {
                const float x1 = bx[j*4], y1 = bx[j*4+1], x2 = bx[j*4+2], y2 = bx[j*4+3];
                const float xx1 = fmaxf(x1, bb0), yy1 = fmaxf(y1, bb1);
                const float xx2 = fminf(x2, bb2), yy2 = fminf(y2, bb3);
                const float inter = fmaxf(0.f, xx2 - xx1 + 1.f) * fmaxf(0.f, yy2 - yy1 + 1.f);
                const float area  = (x2 - x1 + 1.f) * (y2 - y1 + 1.f);
                const float iou   = inter / (area + barea - inter);
                if (iou > 0.7f || j == jm) sc[j] = NEG_BIG;
            }
        }
        __threadfence_block();
        __syncthreads();
    }
}

// ======================================================================
extern "C" void kernel_launch(void* const* d_in, const int* in_sizes, int n_in,
                              void* d_out, int out_size, void* d_ws, size_t ws_size,
                              hipStream_t stream) {
    (void)in_sizes; (void)n_in; (void)out_size; (void)ws_size;
    const float* base_feat = (const float*)d_in[0];
    const float* im_info   = (const float*)d_in[1];
    const float* W_conv    = (const float*)d_in[4];
    const float* b_conv    = (const float*)d_in[5];
    const float* W_cls     = (const float*)d_in[6];
    const float* b_cls     = (const float*)d_in[7];
    const float* W_bbox    = (const float*)d_in[8];
    const float* b_bbox    = (const float*)d_in[9];
    float* out = (float*)d_out;

    char* ws = (char*)d_ws;
    u16t*  feat_bf = (u16t*)(ws + OFF_FEAT);
    u16t*  wgt_bf  = (u16t*)(ws + OFF_WGT);
    u16t*  x_bf    = (u16t*)(ws + OFF_X);
    u16t*  hwgt_bf = (u16t*)(ws + OFF_HWGT);
    float* hout    = (float*)(ws + OFF_HOUT);
    float* props   = (float*)(ws + OFF_PROPS);
    float* scores  = (float*)(ws + OFF_SCORES);
    int*   hist    = (int*)(ws + OFF_HIST);
    int*   cnt     = (int*)(ws + OFF_CNT);
    float* sels    = (float*)(ws + OFF_SELS);
    float* selb    = (float*)(ws + OFF_SELB);

    k_init_select<<<dim3(384), dim3(256), 0, stream>>>(sels, selb, hist, cnt);
    k_repack_feat<<<dim3(HW/32, CI/32, BATCH), dim3(32, 32), 0, stream>>>(base_feat, feat_bf);
    k_repack_wgt<<<dim3(9*CO*CI/256), dim3(256), 0, stream>>>(W_conv, wgt_bf);
    k_repack_hwgt<<<dim3(HEADN*CO/256), dim3(256), 0, stream>>>(W_cls, W_bbox, hwgt_bf);
    k_conv3x3_wmma<<<dim3(CO/64, MTOT/256), dim3(256), 0, stream>>>(feat_bf, wgt_bf, b_conv, x_bf);
    k_head_wmma<<<dim3(MTOT/128), dim3(256), 0, stream>>>(x_bf, hwgt_bf, b_cls, b_bbox, hout);
    k_prep<<<dim3(BATCH*NPROP/256), dim3(256), 0, stream>>>(hout, im_info, props, scores);
    k_hist<<<dim3(BATCH*NPROP/256), dim3(256), 0, stream>>>(scores, hist);
    k_thresh<<<dim3(BATCH), dim3(32), 0, stream>>>(hist, cnt);
    k_compact<<<dim3(BATCH*NPROP/256), dim3(256), 0, stream>>>(scores, props, cnt, sels, selb);
    k_nms<<<dim3(BATCH), dim3(1024), 0, stream>>>(sels, selb, out);
}
